// MultiHeadAttention_77713138253881
// MI455X (gfx1250) — matrix operand order repaired and checked by reading, not compiled, not run on hardware
//
#include <hip/hip_runtime.h>
#include <hip/hip_bf16.h>
#include <hip/hip_fp16.h>

#define B_ 2
#define T_ 2048
#define C_ 1024
#define H_ 16
#define DH_ 64
#define M_ (B_ * T_)   // 4096 rows of x

typedef _Float16 v16h __attribute__((ext_vector_type(16)));
typedef _Float16 v8h  __attribute__((ext_vector_type(8)));
typedef float    v8f  __attribute__((ext_vector_type(8)));

#define WMMA_F16(a, b, c) \
  __builtin_amdgcn_wmma_f32_16x16x32_f16(false, (a), false, (b), (short)0, (c), false, false)

__device__ __forceinline__ v16h cat8(v8h a, v8h b) {
  v16h o;
#pragma unroll
  for (int i = 0; i < 8; ++i) { o[i] = a[i]; o[i + 8] = b[i]; }
  return o;
}

// A fragment: 16x32 f16 tile (M x K), row-major, row stride ld (elements).
// lane m = l&15 holds row m; kg = l>>4 selects K chunks {kg*8..+7, 16+kg*8..+7}.
__device__ __forceinline__ v16h ld_a16x32(const _Float16* p, int ld) {
  const int l = threadIdx.x & 31;
  const int m = l & 15, kg = l >> 4;
  const _Float16* r = p + m * ld + kg * 8;
  v8h a0 = *(const v8h*)(r);
  v8h a1 = *(const v8h*)(r + 16);
  return cat8(a0, a1);
}

// B fragment: 32x16 f16 (K x N). Memory holds B^T rows (each "row" = one
// column of B, K contiguous), row stride ld. lane n = l&15 picks column,
// kg = l>>4 selects K chunks {kg*8..+7, 16+kg*8..+7}.
__device__ __forceinline__ v16h ld_b32x16(const _Float16* p, int ld) {
  const int l = threadIdx.x & 31;
  const int n = l & 15, kg = l >> 4;
  const _Float16* r = p + n * ld + kg * 8;
  v8h b0 = *(const v8h*)(r);
  v8h b1 = *(const v8h*)(r + 16);
  return cat8(b0, b1);
}

// ---------------------------------------------------------------- cast f32->f16
// n must be a multiple of 8*blockDim (sizes here are 4M / 1M elements).
__global__ void cast_f16_kernel(const float* __restrict__ in,
                                _Float16* __restrict__ out, int n) {
  const int i = (blockIdx.x * blockDim.x + threadIdx.x) * 8;
  if (i < n) {
    const float4 f0 = *(const float4*)(in + i);
    const float4 f1 = *(const float4*)(in + i + 4);
    v8h h;
    h[0] = (_Float16)f0.x; h[1] = (_Float16)f0.y;
    h[2] = (_Float16)f0.z; h[3] = (_Float16)f0.w;
    h[4] = (_Float16)f1.x; h[5] = (_Float16)f1.y;
    h[6] = (_Float16)f1.z; h[7] = (_Float16)f1.w;
    *(v8h*)(out + i) = h;
  }
}

// ---------------------------------------------------------------- generic GEMM
// y[row, col] = sum_k A[row,k] * W[col,k]. Each wave: 32(M) x 64(N) tile,
// 8 WMMAs per 32-K step, software-pipelined (double-buffered fragments).
// mode 0/1: f16 out -> Q/K layout [B,H,T,dh]
// mode 2:   f16 out -> Vt layout  [B,H,dh,T]
// mode 3:   f32 out + bias -> d_out [B,T,C]
__global__ __launch_bounds__(256) void gemm_wmma_kernel(
    const _Float16* __restrict__ A, const _Float16* __restrict__ W,
    _Float16* __restrict__ outh, float* __restrict__ outf,
    const float* __restrict__ bias, int mode) {
  const int K = C_, N = C_;
  const int wid = (blockIdx.x * blockDim.x + threadIdx.x) >> 5;
  const int mt  = wid >> 4;   // 0..127  (32-row tile)
  const int nt4 = wid & 15;   // 0..15   (64-col super-tile)
  const int l = threadIdx.x & 31;
  const int n = l & 15, mbase = (l >> 4) * 8;

  v8f acc[2][4] = {};
  const _Float16* Arow0 = A + (size_t)(mt * 32) * K;
  const _Float16* Arow1 = Arow0 + (size_t)16 * K;
  const _Float16* Wb    = W + (size_t)nt4 * 64 * K;

  // prologue: fragments for k = 0
  v16h a0 = ld_a16x32(Arow0, K);
  v16h a1 = ld_a16x32(Arow1, K);
  v16h b0 = ld_b32x16(Wb + (size_t)0 * 16 * K, K);
  v16h b1 = ld_b32x16(Wb + (size_t)1 * 16 * K, K);
  v16h b2 = ld_b32x16(Wb + (size_t)2 * 16 * K, K);
  v16h b3 = ld_b32x16(Wb + (size_t)3 * 16 * K, K);

  for (int k = 0; k < K - 32; k += 32) {
    const int kn = k + 32;
    // stream next tiles (prefetch far ahead: gfx1250 global_prefetch_b8)
    __builtin_prefetch(Arow0 + k + 256, 0, 1);
    __builtin_prefetch(Arow1 + k + 256, 0, 1);
    // issue next-iteration fragment loads before this iteration's WMMAs
    v16h na0 = ld_a16x32(Arow0 + kn, K);
    v16h na1 = ld_a16x32(Arow1 + kn, K);
    v16h nb0 = ld_b32x16(Wb + (size_t)0 * 16 * K + kn, K);
    v16h nb1 = ld_b32x16(Wb + (size_t)1 * 16 * K + kn, K);
    v16h nb2 = ld_b32x16(Wb + (size_t)2 * 16 * K + kn, K);
    v16h nb3 = ld_b32x16(Wb + (size_t)3 * 16 * K + kn, K);

    acc[0][0] = WMMA_F16(a0, b0, acc[0][0]);
    acc[1][0] = WMMA_F16(a1, b0, acc[1][0]);
    acc[0][1] = WMMA_F16(a0, b1, acc[0][1]);
    acc[1][1] = WMMA_F16(a1, b1, acc[1][1]);
    acc[0][2] = WMMA_F16(a0, b2, acc[0][2]);
    acc[1][2] = WMMA_F16(a1, b2, acc[1][2]);
    acc[0][3] = WMMA_F16(a0, b3, acc[0][3]);
    acc[1][3] = WMMA_F16(a1, b3, acc[1][3]);

    a0 = na0; a1 = na1; b0 = nb0; b1 = nb1; b2 = nb2; b3 = nb3;
  }
  // epilogue k-step
  acc[0][0] = WMMA_F16(a0, b0, acc[0][0]);
  acc[1][0] = WMMA_F16(a1, b0, acc[1][0]);
  acc[0][1] = WMMA_F16(a0, b1, acc[0][1]);
  acc[1][1] = WMMA_F16(a1, b1, acc[1][1]);
  acc[0][2] = WMMA_F16(a0, b2, acc[0][2]);
  acc[1][2] = WMMA_F16(a1, b2, acc[1][2]);
  acc[0][3] = WMMA_F16(a0, b3, acc[0][3]);
  acc[1][3] = WMMA_F16(a1, b3, acc[1][3]);

#pragma unroll
  for (int half = 0; half < 2; ++half) {
#pragma unroll
    for (int d = 0; d < 4; ++d) {
      const int col = nt4 * 64 + d * 16 + n;
      float bv = 0.0f;
      if (mode == 3) bv = bias[col];
#pragma unroll
      for (int r = 0; r < 8; ++r) {
        const int row = mt * 32 + half * 16 + mbase + r;
        const float v = acc[half][d][r];
        if (mode == 3) {
          outf[(size_t)row * N + col] = v + bv;
        } else {
          const int b  = row >> 11;       // row / T
          const int t  = row & (T_ - 1);
          const int h  = col >> 6;        // col / dh
          const int dd = col & 63;
          if (mode == 2)
            outh[(((size_t)(b * H_ + h)) * DH_ + dd) * T_ + t] = (_Float16)v;
          else
            outh[(((size_t)(b * H_ + h)) * T_ + t) * DH_ + dd] = (_Float16)v;
        }
      }
    }
  }
}

// ---------------------------------------------------------------- flash attention
// One 32-thread workgroup (= one wave) per 16-query tile. Single-wave
// workgroups: barriers degrade to S_NOP (ISA §3.1) but still order LDS.
__global__ __launch_bounds__(32) void flash_attn_kernel(
    const _Float16* __restrict__ Qh, const _Float16* __restrict__ Kh,
    const _Float16* __restrict__ Vt, _Float16* __restrict__ Oh) {
  __shared__ _Float16 lp[16 * 40];  // P tile 16x32, stride 40 to pad banks

  const int bid = blockIdx.x;
  const int qt = bid & 127;           // T/16 = 128
  const int hh = (bid >> 7) & (H_ - 1);
  const int bb = bid >> 11;
  const int qbase = qt * 16;

  const int l = threadIdx.x;
  const int n = l & 15, kg = l >> 4, mbase = kg * 8;

  const _Float16* qp = Qh + (((size_t)(bb * H_ + hh)) * T_ + qbase) * DH_;
  const _Float16* kp = Kh + (((size_t)(bb * H_ + hh)) * T_) * DH_;
  const _Float16* vp = Vt + (((size_t)(bb * H_ + hh)) * DH_) * T_;

  v16h qa0 = ld_a16x32(qp, DH_);       // head dims 0..31
  v16h qa1 = ld_a16x32(qp + 32, DH_);  // head dims 32..63

  v8f o[4] = {};
  float mrow[8], lrow[8];
#pragma unroll
  for (int r = 0; r < 8; ++r) { mrow[r] = -1e30f; lrow[r] = 0.0f; }

  const int nkb = (qbase + 47) >> 5;   // ceil((qbase+16)/32) key blocks of 32

  for (int kb = 0; kb < nkb; ++kb) {
    const int kb0 = kb * 32;
    // K fragments first, then V fragments: same-type loads complete in
    // order, so S WMMAs need only a partial wait while V streams behind
    // and the softmax VALU stretch hides the rest of the V latency.
    v16h kf0 = ld_b32x16(kp + (size_t)kb0 * DH_, DH_);
    v16h kf1 = ld_b32x16(kp + (size_t)kb0 * DH_ + 32, DH_);
    v16h kf2 = ld_b32x16(kp + (size_t)(kb0 + 16) * DH_, DH_);
    v16h kf3 = ld_b32x16(kp + (size_t)(kb0 + 16) * DH_ + 32, DH_);
    v16h vf[4];
#pragma unroll
    for (int d = 0; d < 4; ++d)
      vf[d] = ld_b32x16(vp + (size_t)(d * 16) * T_ + kb0, T_);
    // stream next key/value block into cache
    __builtin_prefetch(kp + (size_t)(kb0 + 32) * DH_, 0, 1);
    __builtin_prefetch(vp + kb0 + 32, 0, 1);

    v8f s0 = {}, s1 = {};
    s0 = WMMA_F16(qa0, kf0, s0);
    s0 = WMMA_F16(qa1, kf1, s0);
    s1 = WMMA_F16(qa0, kf2, s1);
    s1 = WMMA_F16(qa1, kf3, s1);

    const bool needmask = (kb0 + 31 > qbase);
    const int key0 = kb0 + n, key1 = kb0 + 16 + n;
#pragma unroll
    for (int r = 0; r < 8; ++r) {
      const int q = qbase + mbase + r;
      float a = s0[r] * 0.125f;      // 1/sqrt(64)
      float c = s1[r] * 0.125f;
      if (needmask) {
        if (key0 > q) a = -1e9f;
        if (key1 > q) c = -1e9f;
      }
      s0[r] = a; s1[r] = c;
    }
    // row max (reduce across the 16 lanes of each half; xor masks <=8 stay in-half)
    float rm[8];
#pragma unroll
    for (int r = 0; r < 8; ++r) rm[r] = fmaxf(s0[r], s1[r]);
#pragma unroll
    for (int off = 1; off <= 8; off <<= 1)
#pragma unroll
      for (int r = 0; r < 8; ++r) rm[r] = fmaxf(rm[r], __shfl_xor(rm[r], off, 32));

    float scale[8];
#pragma unroll
    for (int r = 0; r < 8; ++r) {
      const float mn = fmaxf(mrow[r], rm[r]);
      scale[r] = __expf(mrow[r] - mn);
      mrow[r] = mn;
    }
    // P = exp(S - m), row sums
    float rs[8];
#pragma unroll
    for (int r = 0; r < 8; ++r) {
      s0[r] = __expf(s0[r] - mrow[r]);
      s1[r] = __expf(s1[r] - mrow[r]);
      rs[r] = s0[r] + s1[r];
    }
#pragma unroll
    for (int off = 1; off <= 8; off <<= 1)
#pragma unroll
      for (int r = 0; r < 8; ++r) rs[r] += __shfl_xor(rs[r], off, 32);
#pragma unroll
    for (int r = 0; r < 8; ++r) lrow[r] = lrow[r] * scale[r] + rs[r];
#pragma unroll
    for (int d = 0; d < 4; ++d)
#pragma unroll
      for (int r = 0; r < 8; ++r) o[d][r] *= scale[r];

    // D-layout -> A-layout reshape of P through LDS
#pragma unroll
    for (int r = 0; r < 8; ++r) {
      lp[(mbase + r) * 40 + n]      = (_Float16)s0[r];
      lp[(mbase + r) * 40 + 16 + n] = (_Float16)s1[r];
    }
    __syncthreads();
    v16h pa;
    {
      const _Float16* pr = lp + (l & 15) * 40 + kg * 8;
      v8h a0 = *(const v8h*)(pr);
      v8h a1 = *(const v8h*)(pr + 16);
      pa = cat8(a0, a1);
    }
    // O += P (16x32) * V (32x64); Vt rows are V columns, K contiguous
#pragma unroll
    for (int d = 0; d < 4; ++d)
      o[d] = WMMA_F16(pa, vf[d], o[d]);
    __syncthreads();
  }

  float inv[8];
#pragma unroll
  for (int r = 0; r < 8; ++r) inv[r] = 1.0f / lrow[r];
#pragma unroll
  for (int d = 0; d < 4; ++d)
#pragma unroll
    for (int r = 0; r < 8; ++r) {
      const int q = qbase + mbase + r;
      const int col = hh * DH_ + d * 16 + n;
      Oh[((size_t)(bb * T_) + q) * C_ + col] = (_Float16)(o[d][r] * inv[r]);
    }
}

// ---------------------------------------------------------------- launcher
extern "C" void kernel_launch(void* const* d_in, const int* in_sizes, int n_in,
                              void* d_out, int out_size, void* d_ws, size_t ws_size,
                              hipStream_t stream) {
  const float* x  = (const float*)d_in[0];
  const float* Wq = (const float*)d_in[1];
  const float* Wk = (const float*)d_in[2];
  const float* Wv = (const float*)d_in[3];
  const float* Wc = (const float*)d_in[4];
  const float* bc = (const float*)d_in[5];
  float* out = (float*)d_out;

  const size_t NX = (size_t)M_ * C_;   // 4,194,304 activations
  const size_t NW = (size_t)C_ * C_;   // 1,048,576 per weight

  _Float16* xh  = (_Float16*)d_ws;
  _Float16* wqh = xh + NX;
  _Float16* wkh = wqh + NW;
  _Float16* wvh = wkh + NW;
  _Float16* wch = wvh + NW;
  _Float16* qh  = wch + NW;
  _Float16* kh  = qh + NX;
  _Float16* vt  = kh + NX;
  _Float16* oh  = vt + NX;

  // 1) casts (8 elements/thread, b128 in / b128 out)
  cast_f16_kernel<<<(int)(NX / (256 * 8)), 256, 0, stream>>>(x,  xh,  (int)NX);
  cast_f16_kernel<<<(int)(NW / (256 * 8)), 256, 0, stream>>>(Wq, wqh, (int)NW);
  cast_f16_kernel<<<(int)(NW / (256 * 8)), 256, 0, stream>>>(Wk, wkh, (int)NW);
  cast_f16_kernel<<<(int)(NW / (256 * 8)), 256, 0, stream>>>(Wv, wvh, (int)NW);
  cast_f16_kernel<<<(int)(NW / (256 * 8)), 256, 0, stream>>>(Wc, wch, (int)NW);

  // 2) Q/K/V projections: (4096/32) x (1024/64) = 2048 waves = 256 blocks
  const int gemm_grid = (M_ / 32) * (C_ / 64) / 8;  // 256
  gemm_wmma_kernel<<<gemm_grid, 256, 0, stream>>>(xh, wqh, qh, nullptr, nullptr, 0);
  gemm_wmma_kernel<<<gemm_grid, 256, 0, stream>>>(xh, wkh, kh, nullptr, nullptr, 1);
  gemm_wmma_kernel<<<gemm_grid, 256, 0, stream>>>(xh, wvh, vt, nullptr, nullptr, 2);

  // 3) flash attention: one wave per 16-query tile
  flash_attn_kernel<<<B_ * H_ * (T_ / 16), 32, 0, stream>>>(qh, kh, vt, oh);

  // 4) output projection + bias -> f32
  gemm_wmma_kernel<<<gemm_grid, 256, 0, stream>>>(oh, wch, nullptr, out, bc, 3);
}
